// TransformBatchNB_88098369176156
// MI455X (gfx1250) — matrix operand order repaired and checked by reading, not compiled, not run on hardware
//
#include <hip/hip_runtime.h>
#include <hip/hip_bf16.h>

#define BATCH 16
#define NPTS  4096

typedef __attribute__((ext_vector_type(16))) __bf16 v16bf;
typedef __attribute__((ext_vector_type(8)))  __bf16 v8bf;
typedef __attribute__((ext_vector_type(8)))  float  v8f;
typedef __attribute__((ext_vector_type(4)))  unsigned int v4u;
typedef __attribute__((ext_vector_type(4)))  int v4i;
typedef __attribute__((ext_vector_type(8)))  int v8i;

#if defined(__has_builtin)
#  if __has_builtin(__builtin_amdgcn_tensor_load_to_lds) && __has_builtin(__builtin_amdgcn_s_wait_tensorcnt)
#    define HAVE_TDM 1
#  endif
#endif
#ifndef HAVE_TDM
#  define HAVE_TDM 0
#endif

// float atomic max via int/uint trick (works for all signs; init to -inf)
__device__ inline void atomicMaxF(float* addr, float v) {
    if (v >= 0.0f) atomicMax((int*)addr, __float_as_int(v));
    else           atomicMin((unsigned int*)addr, __float_as_uint(v));
}

// ---------------------------------------------------------------------------
// Async-stage a [rows][64] bf16 tile (row stride 64) from global into LDS via
// the Tensor Data Mover (2D D#: tile 64 x rows, data_size=2B, stride 64).
// Entire workgroup must call; one wave issues, all waves sync after.
// ---------------------------------------------------------------------------
__device__ inline void tdm_stage_64ch(const __bf16* __restrict__ gsrc,
                                      __bf16* __restrict__ ldsDst,
                                      int rows, int tid)
{
#if HAVE_TDM
    if (tid < 32) {
        unsigned long long ga = (unsigned long long)(const void*)gsrc;
        unsigned lds_off = (unsigned)(unsigned long long)(void*)ldsDst;
        // D# group 0: count=1 | lds_addr | global_addr | type=2
        v4u g0 = { 1u,
                   lds_off,
                   (unsigned)ga,
                   (unsigned)((ga >> 32) & 0x1FFFFFFu) | (2u << 30) };
        // D# group 1: data_size=2B; tensor_dim0=64; tensor_dim1=rows;
        //             tile_dim0=64; tile_dim1=rows; tensor_dim0_stride=64
        v8i g1 = { (int)(1u << 16),            // [17:16] data_size=1 (2 bytes)
                   (int)(64u << 16),           // bits 63:48 = tensor_dim0 lo16
                   (int)((unsigned)rows << 16),// bits 95:80 = tensor_dim1 lo16
                   (int)(64u << 16),           // bits 127:112 = tile_dim0
                   rows,                       // bits 143:128 = tile_dim1
                   64,                         // bits 191:160 = tensor_dim0_stride lo32
                   0, 0 };
        v4i gz = {};
#if __clang_major__ >= 23
        v8i gz8 = {};
        __builtin_amdgcn_tensor_load_to_lds(g0, g1, gz, gz, gz8, 0);
#else
        __builtin_amdgcn_tensor_load_to_lds(g0, g1, gz, gz, 0);
#endif
        __builtin_amdgcn_s_wait_tensorcnt(0);
    }
    __syncthreads();
#else
    const int n = rows * 64;
    for (int i = tid * 8; i < n; i += 256 * 8)
        *(v8bf*)(ldsDst + i) = *(const v8bf*)(gsrc + i);
    __syncthreads();
#endif
}

// ---------------------------------------------------------------------------
// One WMMA conv layer over a 128-point tile:
//   actOut[p][m] = act( sum_k wgt[m][k] * actIn[p][k] + bias[m] )
// A fragments are hoisted into registers per out-channel tile and reused
// across the wave's PTW point tiles (4 when COUT = 64, 8 when COUT = 128).
// A-fragment (16x32 bf16): lane m=l&15, hi=l>>4,
//   elem i -> K = kb + hi*16 + i
// B-fragment (32x16 bf16): lane col=l&15, elem i -> K = kb + hi*16 + i
// D (16x16 f32): lane col=l&15, vgpr r -> M = r + hi*8
// ---------------------------------------------------------------------------
template<int CIN, int COUT, bool RELU, bool HASBIAS>
__device__ inline void wmma_layer(const __bf16* __restrict__ wgt,    // [COUT][CIN]
                                  const float*  __restrict__ bias,   // [COUT] (or null)
                                  const __bf16* __restrict__ actIn,  // [128][CIN]
                                  __bf16*       __restrict__ actOut, // [128][COUT] (LDS)
                                  int lane, int wave)
{
    constexpr int OCT = COUT / 16;              // out-channel tiles
    constexpr int KT  = CIN / 32;               // k steps
    constexpr int OCW = (OCT >= 8) ? 8 : OCT;   // oc tiles covered per sweep
    constexpr int PTW = 8 / (8 / OCW);          // pt tiles per wave
    const int col = lane & 15;
    const int hi  = lane >> 4;
    const int ocStart = wave % OCW;
    const int pt0 = (wave / OCW) * PTW;

    for (int oc = ocStart; oc < OCT; oc += OCW) {
        const int ocb = oc << 4;
        union { v16bf v; v8bf h[2]; } aF[KT];
        const __bf16* arow = wgt + (size_t)(ocb + col) * CIN;
#pragma unroll
        for (int k = 0; k < KT; ++k) {
            aF[k].h[0] = *(const v8bf*)(arow + 32 * k + hi * 16);
            aF[k].h[1] = *(const v8bf*)(arow + 32 * k + hi * 16 + 8);
        }
#pragma unroll
        for (int pt = pt0; pt < pt0 + PTW; ++pt) {
            const int ptb = pt << 4;
            v8f acc = {};
#pragma unroll
            for (int k = 0; k < KT; ++k) {
                v16bf bf = *(const v16bf*)(actIn + (size_t)(ptb + col) * CIN + 32 * k + hi * 16);
                acc = __builtin_amdgcn_wmma_f32_16x16x32_bf16(
                        false, aF[k].v, false, bf, (short)0, acc, false, false);
            }
#pragma unroll
            for (int r = 0; r < 8; ++r) {
                const int m = ocb + r + hi * 8;
                float v = acc[r];
                if (HASBIAS) v += bias[m];
                if (RELU)    v = fmaxf(v, 0.0f);
                actOut[(ptb + col) * COUT + m] = (__bf16)v;
            }
        }
    }
}

// Final 1024-channel layer fused with masked max. Max is accumulated across
// all 8 point tiles in registers, reduced across 16 lanes once per oc tile,
// then one atomic per (oc tile, r). Bias/relu applied post-max (monotone).
template<int CIN, bool RELU>
__device__ inline void wmma_final(const __bf16* __restrict__ wgt,   // [1024][CIN]
                                  const float*  __restrict__ bias,  // [1024]
                                  const __bf16* __restrict__ actIn, // [128][CIN]
                                  float* __restrict__ gout,         // [1024], this batch
                                  int validCount, int lane, int wave)
{
    constexpr int KT = CIN / 32;
    const int col = lane & 15;
    const int hi  = lane >> 4;
    for (int oc = wave; oc < 64; oc += 8) {
        const int ocb = oc << 4;
        union { v16bf v; v8bf h[2]; } aF[KT];
        const __bf16* arow = wgt + (size_t)(ocb + col) * CIN;
#pragma unroll
        for (int k = 0; k < KT; ++k) {
            aF[k].h[0] = *(const v8bf*)(arow + 32 * k + hi * 16);
            aF[k].h[1] = *(const v8bf*)(arow + 32 * k + hi * 16 + 8);
        }
        float vmax[8];
#pragma unroll
        for (int r = 0; r < 8; ++r) vmax[r] = -__builtin_inff();
#pragma unroll
        for (int pt = 0; pt < 8; ++pt) {
            const int ptb = pt << 4;
            v8f acc = {};
#pragma unroll
            for (int k = 0; k < KT; ++k) {
                v16bf bf = *(const v16bf*)(actIn + (size_t)(ptb + col) * CIN + 32 * k + hi * 16);
                acc = __builtin_amdgcn_wmma_f32_16x16x32_bf16(
                        false, aF[k].v, false, bf, (short)0, acc, false, false);
            }
            const bool valid = (ptb + col) < validCount;
#pragma unroll
            for (int r = 0; r < 8; ++r)
                vmax[r] = fmaxf(vmax[r], valid ? acc[r] : -__builtin_inff());
        }
#pragma unroll
        for (int r = 0; r < 8; ++r) {
            float v = vmax[r];
#pragma unroll
            for (int s = 1; s < 16; s <<= 1) v = fmaxf(v, __shfl_xor(v, s));
            v += bias[ocb + r + hi * 8];
            if (RELU) v = fmaxf(v, 0.0f);
            if (col == 0) atomicMaxF(&gout[ocb + r + hi * 8], v);
        }
    }
}

// ---------------------------------------------------------------------------
__global__ void k_fill(float* __restrict__ p, float v, int n) {
    int i = blockIdx.x * blockDim.x + threadIdx.x;
    if (i < n) p[i] = v;
}

__global__ void k_cvt_bf16(const float* __restrict__ s, __bf16* __restrict__ d, int n) {
    int i = blockIdx.x * blockDim.x + threadIdx.x;
    if (i < n) d[i] = (__bf16)s[i];
}

// tnet3 conv stack: 3 -> 64 (VALU) -> 128 (WMMA) -> 1024 (WMMA) + masked max
__global__ __launch_bounds__(256)
void k_tnet3_stack(const float* __restrict__ points, const int* __restrict__ lengths,
                   const float*  __restrict__ c1w, const float* __restrict__ c1b,
                   const __bf16* __restrict__ c2w, const float* __restrict__ c2b,
                   const __bf16* __restrict__ c3w, const float* __restrict__ c3b,
                   float* __restrict__ g3)
{
    __shared__ float  spts[128 * 3];
    __shared__ __bf16 act1[128 * 64];
    __shared__ __bf16 act2[128 * 128];
    const int b = blockIdx.y;
    const int ptBase = blockIdx.x * 128;
    const int tid = threadIdx.x, lane = tid & 31, wave = tid >> 5;

    for (int i = tid; i < 128 * 3; i += 256) {
        int k = i / 128, p = i % 128;
        spts[p * 3 + k] = points[((size_t)b * 3 + k) * NPTS + ptBase + p];
    }
    __syncthreads();
    for (int i = tid; i < 128 * 64; i += 256) {
        int p = i >> 6, c = i & 63;
        float s = c1b[c] + c1w[c * 3] * spts[p * 3]
                + c1w[c * 3 + 1] * spts[p * 3 + 1]
                + c1w[c * 3 + 2] * spts[p * 3 + 2];
        act1[p * 64 + c] = (__bf16)fmaxf(s, 0.0f);
    }
    __syncthreads();
    wmma_layer<64, 128, true, true>(c2w, c2b, act1, act2, lane, wave);
    __syncthreads();
    const int vc = min(max(lengths[b] - ptBase, 0), 128);
    wmma_final<128, true>(c3w, c3b, act2, g3 + b * 1024, vc, lane, wave);
}

// FC head of a T-Net: g[1024] -> relu 512 -> relu 256 -> k*k (+ I)
__global__ __launch_bounds__(256)
void k_fc_head(const float* __restrict__ g,
               const float* __restrict__ f1w, const float* __restrict__ f1b,
               const float* __restrict__ f2w, const float* __restrict__ f2b,
               const float* __restrict__ f3w, const float* __restrict__ f3b,
               float* __restrict__ mOut, int k)
{
    __shared__ float s0[1024];
    __shared__ float s1[512];
    __shared__ float s2[256];
    const int b = blockIdx.x, tid = threadIdx.x;
    for (int i = tid; i < 1024; i += 256) s0[i] = g[b * 1024 + i];
    __syncthreads();
    for (int o = tid; o < 512; o += 256) {
        float s = f1b[o];
        const float* w = f1w + (size_t)o * 1024;
        for (int i = 0; i < 1024; ++i) s += w[i] * s0[i];
        s1[o] = fmaxf(s, 0.0f);
    }
    __syncthreads();
    if (tid < 256) {
        float s = f2b[tid];
        const float* w = f2w + (size_t)tid * 512;
        for (int i = 0; i < 512; ++i) s += w[i] * s1[i];
        s2[tid] = fmaxf(s, 0.0f);
    }
    __syncthreads();
    const int kk = k * k;
    for (int o = tid; o < kk; o += 256) {
        float s = f3b[o];
        const float* w = f3w + (size_t)o * 256;
        for (int i = 0; i < 256; ++i) s += w[i] * s2[i];
        if ((o / k) == (o % k)) s += 1.0f;
        mOut[(size_t)b * kk + o] = s;
    }
}

// x1[b][n][o] = relu( c1( m3-transform(points) ) )   (point-major bf16)
__global__ __launch_bounds__(256)
void k_transform_c1(const float* __restrict__ points, const float* __restrict__ m3,
                    const float* __restrict__ c1w, const float* __restrict__ c1b,
                    __bf16* __restrict__ x1)
{
    __shared__ float wt[64 * 3];   // fused weight: wt[o][k] = sum_j c1w[o][j]*m3[b][k][j]
    __shared__ float sp[256 * 3];
    const int b = blockIdx.y, ptBase = blockIdx.x * 256;
    const int tid = threadIdx.x;
    if (tid < 192) {
        int o = tid / 3, k = tid % 3;
        const float* m = m3 + b * 9;
        wt[o * 3 + k] = c1w[o * 3] * m[k * 3] + c1w[o * 3 + 1] * m[k * 3 + 1]
                      + c1w[o * 3 + 2] * m[k * 3 + 2];
    }
    for (int i = tid; i < 256 * 3; i += 256) {
        int k = i / 256, p = i % 256;
        sp[p * 3 + k] = points[((size_t)b * 3 + k) * NPTS + ptBase + p];
    }
    __syncthreads();
    const int p = tid;
    const float a0 = sp[p * 3], a1 = sp[p * 3 + 1], a2 = sp[p * 3 + 2];
    __bf16* dst = x1 + ((size_t)b * NPTS + ptBase + p) * 64;
    for (int o = 0; o < 64; ++o) {
        float v = fmaf(wt[o * 3], a0, fmaf(wt[o * 3 + 1], a1, fmaf(wt[o * 3 + 2], a2, c1b[o])));
        dst[o] = (__bf16)fmaxf(v, 0.0f);
    }
}

// tnet64 conv stack: 64 -> 64 -> 128 -> 1024 + masked max.
// Input tile staged global->LDS with the Tensor Data Mover (async, TENSORcnt).
__global__ __launch_bounds__(256)
void k_tnet64_stack(const __bf16* __restrict__ x1, const int* __restrict__ lengths,
                    const __bf16* __restrict__ w1, const float* __restrict__ b1,
                    const __bf16* __restrict__ w2, const float* __restrict__ b2,
                    const __bf16* __restrict__ w3, const float* __restrict__ b3,
                    float* __restrict__ g64)
{
    __shared__ __bf16 act0[128 * 64];
    __shared__ __bf16 act1[128 * 64];
    __shared__ __bf16 act2[128 * 128];
    const int b = blockIdx.y, ptBase = blockIdx.x * 128;
    const int tid = threadIdx.x, lane = tid & 31, wave = tid >> 5;
    tdm_stage_64ch(x1 + ((size_t)b * NPTS + ptBase) * 64, act0, 128, tid);
    wmma_layer<64, 64, true, true>(w1, b1, act0, act1, lane, wave);
    __syncthreads();
    wmma_layer<64, 128, true, true>(w2, b2, act1, act2, lane, wave);
    __syncthreads();
    const int vc = min(max(lengths[b] - ptBase, 0), 128);
    wmma_final<128, true>(w3, b3, act2, g64 + b * 1024, vc, lane, wave);
}

// final stack: m64-transform (WMMA, A = m64^T) -> c2 -> c3 (+ masked max, no relu)
__global__ __launch_bounds__(256)
void k_final_stack(const __bf16* __restrict__ x1, const int* __restrict__ lengths,
                   const float*  __restrict__ m64,
                   const __bf16* __restrict__ w2, const float* __restrict__ b2,
                   const __bf16* __restrict__ w3, const float* __restrict__ b3,
                   float* __restrict__ outp)
{
    __shared__ __bf16 sm64[64 * 64];   // W'[j][k] = m64[b][k][j]
    __shared__ __bf16 actT[128 * 64];
    __shared__ __bf16 act2[128 * 128];
    const int b = blockIdx.y, ptBase = blockIdx.x * 128;
    const int tid = threadIdx.x, lane = tid & 31, wave = tid >> 5;
    const float* m = m64 + (size_t)b * 4096;
    for (int i = tid; i < 4096; i += 256) {
        int j = i >> 6, k = i & 63;
        sm64[j * 64 + k] = (__bf16)m[k * 64 + j];
    }
    __syncthreads();
    const __bf16* act0 = x1 + ((size_t)b * NPTS + ptBase) * 64;
    wmma_layer<64, 64, false, false>(sm64, nullptr, act0, actT, lane, wave);
    __syncthreads();
    wmma_layer<64, 128, true, true>(w2, b2, actT, act2, lane, wave);
    __syncthreads();
    const int vc = min(max(lengths[b] - ptBase, 0), 128);
    wmma_final<128, false>(w3, b3, act2, outp + b * 1024, vc, lane, wave);
}

// ---------------------------------------------------------------------------
extern "C" void kernel_launch(void* const* d_in, const int* in_sizes, int n_in,
                              void* d_out, int out_size, void* d_ws, size_t ws_size,
                              hipStream_t stream)
{
    const float* points = (const float*)d_in[0];
    const int*   lengths = (const int*)d_in[1];
    // tnet3 params
    const float* t3c1w = (const float*)d_in[2];
    const float* t3c1b = (const float*)d_in[3];
    const float* t3c2w = (const float*)d_in[4];
    const float* t3c2b = (const float*)d_in[5];
    const float* t3c3w = (const float*)d_in[6];
    const float* t3c3b = (const float*)d_in[7];
    const float* t3f1w = (const float*)d_in[8];
    const float* t3f1b = (const float*)d_in[9];
    const float* t3f2w = (const float*)d_in[10];
    const float* t3f2b = (const float*)d_in[11];
    const float* t3f3w = (const float*)d_in[12];
    const float* t3f3b = (const float*)d_in[13];
    // tnet64 params
    const float* t64c1w = (const float*)d_in[14];
    const float* t64c1b = (const float*)d_in[15];
    const float* t64c2w = (const float*)d_in[16];
    const float* t64c2b = (const float*)d_in[17];
    const float* t64c3w = (const float*)d_in[18];
    const float* t64c3b = (const float*)d_in[19];
    const float* t64f1w = (const float*)d_in[20];
    const float* t64f1b = (const float*)d_in[21];
    const float* t64f2w = (const float*)d_in[22];
    const float* t64f2b = (const float*)d_in[23];
    const float* t64f3w = (const float*)d_in[24];
    const float* t64f3b = (const float*)d_in[25];
    // top-level convs
    const float* c1w = (const float*)d_in[26];
    const float* c1b = (const float*)d_in[27];
    const float* c2w = (const float*)d_in[28];
    const float* c2b = (const float*)d_in[29];
    const float* c3w = (const float*)d_in[30];
    const float* c3b = (const float*)d_in[31];

    float* out = (float*)d_out;      // [16,1024]
    float* m3  = out + 16 * 1024;    // [16,3,3]
    float* m64 = m3 + 16 * 9;        // [16,64,64]

    // workspace layout
    float*  g3  = (float*)d_ws;       // [16,1024]
    float*  g64 = g3 + 16 * 1024;     // [16,1024]
    __bf16* wb  = (__bf16*)(g64 + 16 * 1024);
    __bf16* t3c2wb  = wb; wb += 128 * 64;
    __bf16* t3c3wb  = wb; wb += 1024 * 128;
    __bf16* t64c1wb = wb; wb += 64 * 64;
    __bf16* t64c2wb = wb; wb += 128 * 64;
    __bf16* t64c3wb = wb; wb += 1024 * 128;
    __bf16* c2wb    = wb; wb += 128 * 64;
    __bf16* c3wb    = wb; wb += 1024 * 128;
    __bf16* x1      = wb;             // [16,4096,64] bf16, point-major

    auto cvt = [&](const float* s, __bf16* d, int n) {
        k_cvt_bf16<<<(n + 255) / 256, 256, 0, stream>>>(s, d, n);
    };
    cvt(t3c2w,  t3c2wb,  128 * 64);
    cvt(t3c3w,  t3c3wb,  1024 * 128);
    cvt(t64c1w, t64c1wb, 64 * 64);
    cvt(t64c2w, t64c2wb, 128 * 64);
    cvt(t64c3w, t64c3wb, 1024 * 128);
    cvt(c2w,    c2wb,    128 * 64);
    cvt(c3w,    c3wb,    1024 * 128);

    k_fill<<<64, 256, 0, stream>>>(g3,  -__builtin_inff(), 16 * 1024);
    k_fill<<<64, 256, 0, stream>>>(g64, -__builtin_inff(), 16 * 1024);
    k_fill<<<64, 256, 0, stream>>>(out, -__builtin_inff(), 16 * 1024);

    dim3 gridConv(NPTS / 128, BATCH);
    k_tnet3_stack<<<gridConv, 256, 0, stream>>>(points, lengths,
        t3c1w, t3c1b, t3c2wb, t3c2b, t3c3wb, t3c3b, g3);
    k_fc_head<<<BATCH, 256, 0, stream>>>(g3, t3f1w, t3f1b, t3f2w, t3f2b, t3f3w, t3f3b, m3, 3);

    dim3 gridT(NPTS / 256, BATCH);
    k_transform_c1<<<gridT, 256, 0, stream>>>(points, m3, c1w, c1b, x1);

    k_tnet64_stack<<<gridConv, 256, 0, stream>>>(x1, lengths,
        t64c1wb, t64c1b, t64c2wb, t64c2b, t64c3wb, t64c3b, g64);
    k_fc_head<<<BATCH, 256, 0, stream>>>(g64, t64f1w, t64f1b, t64f2w, t64f2b, t64f3w, t64f3b, m64, 64);

    k_final_stack<<<gridConv, 256, 0, stream>>>(x1, lengths, m64, c2wb, c2b, c3wb, c3b, out);
}
